// QKVAttention_89799176225592
// MI455X (gfx1250) — matrix operand order repaired and checked by reading, not compiled, not run on hardware
//
#include <hip/hip_runtime.h>

typedef _Float16 v16h __attribute__((ext_vector_type(16)));
typedef _Float16 v8h  __attribute__((ext_vector_type(8)));
typedef _Float16 h2   __attribute__((ext_vector_type(2)));
typedef float    v8f  __attribute__((ext_vector_type(8)));
typedef float    v4f  __attribute__((ext_vector_type(4)));

#define B_    8
#define T_    4096
#define D_    64
#define PITCH 192          // qkv row pitch in halves (3*64)
#define KOFF  64           // k channel offset
#define VOFF  128          // v channel offset
#define SBLK  32           // keys per inner iteration
#define QT    32           // query rows per wave (2 sub-tiles of 16)
#define WPB   4            // waves per block (attention kernel)

// lane-half swap (xor 0x10) within wave32 via ds_swizzle SWAPX16
__device__ __forceinline__ float swz16f(float x) {
  int i = __builtin_bit_cast(int, x);
  i = __builtin_amdgcn_ds_swizzle(i, 0x401F);
  return __builtin_bit_cast(float, i);
}
__device__ __forceinline__ h2 swz16h(h2 x) {
  int i = __builtin_bit_cast(int, x);
  i = __builtin_amdgcn_ds_swizzle(i, 0x401F);
  return __builtin_bit_cast(h2, i);
}
// f32 pair -> packed f16 (v_cvt_pk_f16_f32)
__device__ __forceinline__ h2 pk16(float a, float b) {
  return __builtin_bit_cast(h2, __builtin_amdgcn_cvt_pkrtz(a, b));
}

union V16U {
  v16h v;
  v8h  h8[2];
  h2   hp[8];
  _Float16 h[16];
};

// ---------------------------------------------------------------------------
// Kernel 1: transpose V out of the interleaved qkv into d_ws as V^T [B][64][T]
// so the attention kernel can load V^T A-operands with contiguous b128 reads.
// ---------------------------------------------------------------------------
__global__ void __launch_bounds__(256) vt_transpose(const _Float16* __restrict__ qkv,
                                                    _Float16* __restrict__ vt) {
  __shared__ __align__(16) _Float16 tile[64][72];   // 72-half stride keeps 16B align
  const int b  = blockIdx.x >> 6;        // 64 t-tiles per batch
  const int tb = (blockIdx.x & 63) << 6; // 64 rows of t per block
  const int tid = threadIdx.x;
#pragma unroll
  for (int i = 0; i < 2; ++i) {
    int cid = tid + i * 256;             // 512 chunks of 8 halves
    int tl  = cid >> 3;
    int d0  = (cid & 7) << 3;
    v8h src = *(const v8h*)(qkv + ((size_t)(b * T_ + tb + tl)) * PITCH + VOFF + d0);
    *(v8h*)&tile[tl][d0] = src;
  }
  __syncthreads();
#pragma unroll
  for (int i = 0; i < 2; ++i) {
    int cid = tid + i * 256;
    int d   = cid >> 3;
    int t0  = (cid & 7) << 3;
    v8h dst;
#pragma unroll
    for (int j = 0; j < 8; ++j) dst[j] = tile[t0 + j][d];
    *(v8h*)(vt + ((size_t)(b * D_ + d)) * T_ + tb + t0) = dst;
  }
}

// ---------------------------------------------------------------------------
// Kernel 2: precompute scaled k row norms ksq[b][s] = SCALE^2 * sum(k^2),
// shared by all q-tile waves of each head.
// ---------------------------------------------------------------------------
__global__ void __launch_bounds__(256) ksq_kernel(const _Float16* __restrict__ qkv,
                                                  float* __restrict__ ksq) {
  const int idx = blockIdx.x * 256 + threadIdx.x;    // b*T + s, 0..32767
  const _Float16* krow = qkv + (size_t)idx * PITCH + KOFF;
  float ks = 0.f;
#pragma unroll
  for (int j = 0; j < 8; ++j) {
    v8h kv = *(const v8h*)(krow + j * 8);
#pragma unroll
    for (int i = 0; i < 8; ++i) { float a = (float)kv[i]; ks += a * a; }
  }
  ksq[idx] = ks * 0.015625f;                         // fold SCALE^2 = 1/64
}

// ---------------------------------------------------------------------------
// Kernel 3: fused distance-attention. One wave owns 32 query rows (2 sub-tiles)
// of one head, so K/V loads are amortized over 18 WMMAs per 32-key block.
// S^T formulation: matmul-1 computes S^T = K x Q^T, matmul-2 computes
// O^T = V^T x P^T, so P converts C-layout -> B-layout with in-lane pack only.
// ---------------------------------------------------------------------------
__global__ void __launch_bounds__(32 * WPB, 1)
attn_kernel(const _Float16* __restrict__ qkv,
            const _Float16* __restrict__ vt,
            const float* __restrict__ ksq,
            _Float16* __restrict__ out) {
  const int lane  = threadIdx.x & 31;
  const int wave  = threadIdx.x >> 5;
  const int gw    = blockIdx.x * WPB + wave;     // global wave id
  const int b     = gw >> 7;                     // 128 q-tiles (of 32) per head
  const int tbase = (gw & 127) << 5;
  const int hi    = lane >> 4;                   // lane half
  const int lj    = lane & 15;
  const int qc    = hi ? 8 : 0;                  // B layout: lanes<16 K=0-7,16-23
  const int ac    = hi ? 8 : 0;                  // A layout: lanes<16 K=0-7,16-23

  // ---- Q^T B-operands for both sub-tiles, loaded once ----------------------
  V16U bq[2][2];
  float qs[2];
#pragma unroll
  for (int q = 0; q < 2; ++q) {
    const _Float16* qrow = qkv + ((size_t)(b * T_ + tbase + q * 16 + lj)) * PITCH;
    bq[q][0].h8[0] = *(const v8h*)(qrow + qc);
    bq[q][0].h8[1] = *(const v8h*)(qrow + qc + 16);
    bq[q][1].h8[0] = *(const v8h*)(qrow + 32 + qc);
    bq[q][1].h8[1] = *(const v8h*)(qrow + 32 + qc + 16);
    float s = 0.f;
#pragma unroll
    for (int i = 0; i < 16; ++i) { float a = (float)bq[q][0].h[i]; s += a * a; }
#pragma unroll
    for (int i = 0; i < 16; ++i) { float a = (float)bq[q][1].h[i]; s += a * a; }
    qs[q] = (s + swz16f(s)) * 0.015625f;         // q row norm * SCALE^2
  }

  // all-ones A operand for the row-sum WMMA
  V16U ones;
#pragma unroll
  for (int i = 0; i < 16; ++i) ones.h[i] = (_Float16)1.0f;

  v8f oacc[2][4] = {{v8f{}, v8f{}, v8f{}, v8f{}}, {v8f{}, v8f{}, v8f{}, v8f{}}};
  v8f lacc[2] = {v8f{}, v8f{}};

  // strength-reduced bases (advance by pointer increments; sub-offsets are
  // immediate-encodable: g=1 K rows at +16*PITCH halves, V d-groups at +dg*16*T_)
  const _Float16* kp  = qkv + ((size_t)(b * T_ + lj)) * PITCH + KOFF + ac;
  const _Float16* vp  = vt + ((size_t)(b * D_ + lj)) * T_ + ac;
  const float*    kqp = ksq + b * T_ + hi * 8;

#pragma unroll 1
  for (int s0 = 0; s0 < T_; s0 += SBLK) {
    // ---- K A-operands: 2 s-groups x 2 dim-halves (8 x b128) ----------------
    V16U ak[2][2];
#pragma unroll
    for (int g = 0; g < 2; ++g) {
      const _Float16* kr = kp + g * 16 * PITCH;
      ak[g][0].h8[0] = *(const v8h*)(kr);
      ak[g][0].h8[1] = *(const v8h*)(kr + 16);
      ak[g][1].h8[0] = *(const v8h*)(kr + 32);
      ak[g][1].h8[1] = *(const v8h*)(kr + 48);
    }
    // precomputed k norms, indexed by the C-tile M axis: s = s0 + g*16 + 8*hi + r
    float kr0[8], kr1[8];
    *(v4f*)&kr0[0] = *(const v4f*)(kqp);
    *(v4f*)&kr0[4] = *(const v4f*)(kqp + 4);
    *(v4f*)&kr1[0] = *(const v4f*)(kqp + 16);
    *(v4f*)&kr1[4] = *(const v4f*)(kqp + 20);

    // ---- matmul-1: S^T tiles for both q sub-tiles (8 WMMAs) ----------------
    v8f c[2][2] = {{v8f{}, v8f{}}, {v8f{}, v8f{}}};
#pragma unroll
    for (int q = 0; q < 2; ++q) {
#pragma unroll
      for (int g = 0; g < 2; ++g) {
        c[q][g] = __builtin_amdgcn_wmma_f32_16x16x32_f16(
            false, ak[g][0].v, false, bq[q][0].v, (short)0, c[q][g], false, false);
        c[q][g] = __builtin_amdgcn_wmma_f32_16x16x32_f16(
            false, ak[g][1].v, false, bq[q][1].v, (short)0, c[q][g], false, false);
      }
    }

    // ---- elementwise p = exp(-||q-k||), then C->B conversion ---------------
    V16U bp[2];
#pragma unroll
    for (int q = 0; q < 2; ++q) {
#pragma unroll
      for (int r = 0; r < 8; ++r) {
        float d2 = (qs[q] + kr0[r]) - 0.03125f * c[q][0][r];   // 2*SCALE^2=1/32
        float dd = __builtin_amdgcn_sqrtf(fmaxf(d2, 0.f));
        c[q][0][r] = __builtin_amdgcn_exp2f(dd * -1.44269504088896341f);
        d2 = (qs[q] + kr1[r]) - 0.03125f * c[q][1][r];
        dd = __builtin_amdgcn_sqrtf(fmaxf(d2, 0.f));
        c[q][1][r] = __builtin_amdgcn_exp2f(dd * -1.44269504088896341f);
      }
#pragma unroll
      for (int j = 0; j < 4; ++j) {
        h2 p0 = pk16(c[q][0][2 * j], c[q][0][2 * j + 1]);
        h2 p1 = pk16(c[q][1][2 * j], c[q][1][2 * j + 1]);
        h2 sp0 = swz16h(p0);
        h2 sp1 = swz16h(p1);
        bp[q].hp[j]     = p0;              // K = s 0..7   / 8..15
        bp[q].hp[4 + j] = p1;              // K = s 16..23 / 24..31
      }
    }

    // ---- matmul-2: O^T += V^T x P^T (8 WMMAs) + row sums (2 WMMAs) ---------
#pragma unroll
    for (int dg = 0; dg < 4; ++dg) {
      V16U av;
      av.h8[0] = *(const v8h*)(vp + dg * 16 * T_);
      av.h8[1] = *(const v8h*)(vp + dg * 16 * T_ + 16);
      oacc[0][dg] = __builtin_amdgcn_wmma_f32_16x16x32_f16(
          false, av.v, false, bp[0].v, (short)0, oacc[0][dg], false, false);
      oacc[1][dg] = __builtin_amdgcn_wmma_f32_16x16x32_f16(
          false, av.v, false, bp[1].v, (short)0, oacc[1][dg], false, false);
    }
    lacc[0] = __builtin_amdgcn_wmma_f32_16x16x32_f16(
        false, ones.v, false, bp[0].v, (short)0, lacc[0], false, false);
    lacc[1] = __builtin_amdgcn_wmma_f32_16x16x32_f16(
        false, ones.v, false, bp[1].v, (short)0, lacc[1], false, false);

    kp  += SBLK * PITCH;
    vp  += SBLK;
    kqp += SBLK;
  }

  // ---- epilogue: out[t][d] = O^T[d][t] / l[t] ------------------------------
#pragma unroll
  for (int q = 0; q < 2; ++q) {
    float inv = __builtin_amdgcn_rcpf(lacc[q][0]);  // every VGPR of lacc == l[t]
    _Float16* orow = out + ((size_t)(b * T_ + tbase + q * 16 + lj)) * D_ + hi * 8;
#pragma unroll
    for (int dg = 0; dg < 4; ++dg) {
      v8h st;
#pragma unroll
      for (int r = 0; r < 8; ++r) st[r] = (_Float16)(oacc[q][dg][r] * inv);
      *(v8h*)(orow + dg * 16) = st;
    }
  }
}

// ---------------------------------------------------------------------------
extern "C" void kernel_launch(void* const* d_in, const int* in_sizes, int n_in,
                              void* d_out, int out_size, void* d_ws, size_t ws_size,
                              hipStream_t stream) {
  const _Float16* qkv = (const _Float16*)d_in[0];
  _Float16* out = (_Float16*)d_out;
  _Float16* vtw = (_Float16*)d_ws;                       // 4 MB: V^T [B][64][T]
  float* ksqw = (float*)((char*)d_ws + (size_t)B_ * D_ * T_ * sizeof(_Float16));
                                                         // +128 KB: ksq [B][T]

  // 1) V^T into workspace (coalesced LDS tile transpose)
  vt_transpose<<<dim3(B_ * (T_ / 64)), dim3(256), 0, stream>>>(qkv, vtw);

  // 2) scaled k row norms, shared by all waves
  ksq_kernel<<<dim3(B_ * T_ / 256), dim3(256), 0, stream>>>(qkv, ksqw);

  // 3) fused distance attention: 1024 waves (32 q-rows each), 4 per block
  attn_kernel<<<dim3((B_ * T_ / QT) / WPB), dim3(32 * WPB), 0, stream>>>(qkv, vtw, ksqw, out);
}
